// GTOTRegularization_81578608820340
// MI455X (gfx1250) — matrix operand order repaired and checked by reading, not compiled, not run on hardware
//
#include <hip/hip_runtime.h>
#include <hip/hip_bf16.h>

// ---------------------------------------------------------------------------
// GTOT regularization for MI455X (gfx1250), wave32.
//
// Problem: B=256 graphs, n=256 nodes/graph, D=256 features.
//   C_b = 1 - xs_n_b @ xt_n_b^T          (256 GEMMs of 256x256x256)
//   support_b[i][j] = (adj edge) || (i==j)  (only A==0 matters downstream)
//   log-domain Sinkhorn (eps=0.1, <=100 iters) on C_b masked by support
//   wd_b = sum(pi * C);  out = 0.5 * mean_b(wd_b)
//
// Design: one workgroup (256 thr = 8 waves) per graph. C (256x257 f32,
// padded pitch) + bitpacked support (8KB) + u/v live in LDS (~273KB of the
// 320KB/WGP budget), so 100 Sinkhorn iterations never touch HBM. C is built
// with v_wmma_f32_16x16x32_f16 from f16-normalized features (f32 accumulate).
// ---------------------------------------------------------------------------

typedef __attribute__((ext_vector_type(8)))  _Float16 v8h;
typedef __attribute__((ext_vector_type(16))) _Float16 v16h;
typedef __attribute__((ext_vector_type(8)))  float    v8f;

#define N_NODES   65536
#define D_FEAT    256
#define N_GRAPHS  256
#define NPG       256            // nodes per graph
#define N_EDGES   524288
#define SUPP_WPR  8              // 256 bits / 32 = 8 words per row
#define SUPP_WPG  (NPG * SUPP_WPR)   // 2048 words per graph

#define EPS       0.1f
#define INV_EPS   10.0f
#define THRESH    0.1f
#define MAX_ITER  100
#define NEG_INF   (-1.0e9f)

// LDS layout (in floats)
#define C_PITCH   257
#define U_OFF     (NPG * C_PITCH)     // 65792
#define V_OFF     (U_OFF + NPG)       // 66048
#define RED_OFF   (V_OFF + NPG)       // 66304 (8 floats)
#define SUPP_OFF  (RED_OFF + 8)       // 66312 (2048 u32)
#define LDS_FLOATS (SUPP_OFF + SUPP_WPG)
#define LDS_BYTES  (LDS_FLOATS * 4)   // 273440 B  (< 320KB per WG)

__device__ __forceinline__ float wave_sum32(float x) {
  #pragma unroll
  for (int m = 16; m >= 1; m >>= 1) x += __shfl_xor(x, m, 32);
  return x;
}
__device__ __forceinline__ float wave_max32(float x) {
  #pragma unroll
  for (int m = 16; m >= 1; m >>= 1) x = fmaxf(x, __shfl_xor(x, m, 32));
  return x;
}

// ---------------------------------------------------------------------------
// Kernel 1: row-normalize f32 features -> f16 (one wave per row)
// ---------------------------------------------------------------------------
__global__ void gtot_rownorm_f16(const float* __restrict__ in,
                                 _Float16* __restrict__ out) {
  const int lane = threadIdx.x & 31;
  const int wave = threadIdx.x >> 5;
  const int row  = blockIdx.x * 8 + wave;       // blockDim = 256 -> 8 rows/blk
  const float* p = in + (size_t)row * D_FEAT;
  float vals[8];
  float ss = 0.0f;
  #pragma unroll
  for (int q = 0; q < 8; ++q) {
    vals[q] = p[lane + 32 * q];
    ss += vals[q] * vals[q];
  }
  ss = wave_sum32(ss);
  const float inv = 1.0f / (sqrtf(ss) + 1e-12f);
  _Float16* o = out + (size_t)row * D_FEAT;
  #pragma unroll
  for (int q = 0; q < 8; ++q) o[lane + 32 * q] = (_Float16)(vals[q] * inv);
}

// ---------------------------------------------------------------------------
// Kernel 2: init support mask with self-loop diagonal bits
// word index = b*2048 + i*8 + q ; diagonal bit for row i lives in word i>>5
// ---------------------------------------------------------------------------
__global__ void gtot_supp_init(unsigned* __restrict__ supp) {
  const int idx = blockIdx.x * blockDim.x + threadIdx.x;  // < 256*2048
  const int rem = idx & (SUPP_WPG - 1);
  const int i = rem >> 3;
  const int q = rem & 7;
  supp[idx] = (q == (i >> 5)) ? (1u << (i & 31)) : 0u;
}

// ---------------------------------------------------------------------------
// Kernel 3: scatter edges into support mask (atomicOr).
// edge_index is [2, E] flat: u = ei[t], v = ei[E + t].  b = u>>8 (equal-size
// sorted graphs), r = u&255, c = v&255 (edges are intra-graph by construction)
// ---------------------------------------------------------------------------
__global__ void gtot_supp_edges(const int* __restrict__ ei,
                                unsigned* __restrict__ supp) {
  const int t = blockIdx.x * blockDim.x + threadIdx.x;    // < E
  const int u = ei[t];
  const int v = ei[N_EDGES + t];
  const int b = u >> 8;
  const int r = u & 255;
  const int c = v & 255;
  atomicOr(&supp[b * SUPP_WPG + r * SUPP_WPR + (c >> 5)], 1u << (c & 31));
}

// ---------------------------------------------------------------------------
// Kernel 4: per-graph workgroup — WMMA cost GEMM into LDS, Sinkhorn in LDS,
// per-graph OT cost out.
// ---------------------------------------------------------------------------
__global__ void __launch_bounds__(256)
gtot_graph(const _Float16* __restrict__ xs_h,
           const _Float16* __restrict__ xt_h,
           const unsigned* __restrict__ supp_g,
           float* __restrict__ wd_out) {
  extern __shared__ float smem[];
  float*    Cs   = smem;
  float*    u    = smem + U_OFF;
  float*    v    = smem + V_OFF;
  float*    red  = smem + RED_OFF;
  unsigned* supp = (unsigned*)(smem + SUPP_OFF);

  const int b    = blockIdx.x;
  const int tid  = threadIdx.x;
  const int lane = tid & 31;
  const int wave = tid >> 5;
  const int half = lane >> 4;     // 0: lanes 0-15, 1: lanes 16-31
  const int l16  = lane & 15;

  // stage support mask into LDS; init u, v
  for (int w = tid; w < SUPP_WPG; w += 256) supp[w] = supp_g[b * SUPP_WPG + w];
  u[tid] = 0.0f;
  v[tid] = 0.0f;

  // ---- Phase 1: C = 1 - xs_n @ xt_n^T via v_wmma_f32_16x16x32_f16 --------
  const _Float16* XS = xs_h + (size_t)b * NPG * D_FEAT;
  const _Float16* XT = xt_h + (size_t)b * NPG * D_FEAT;

  for (int t = wave; t < 256; t += 8) {          // 16x16 output tiles
    const int i0 = (t >> 4) << 4;
    const int j0 = (t & 15) << 4;
    v8f acc = {0.f, 0.f, 0.f, 0.f, 0.f, 0.f, 0.f, 0.f};
    const _Float16* ap = XS + (size_t)(i0 + l16) * D_FEAT;  // A row (M)
    const _Float16* bp = XT + (size_t)(j0 + l16) * D_FEAT;  // B col (N) = xt row
    #pragma unroll 2
    for (int k0 = 0; k0 < D_FEAT; k0 += 32) {
      // A 16x32 f16: elems 0..7 -> K = k0+8h..+7 ; elems 8..15 -> K = k0+16+8h..
      union { v16h v; v8h h[2]; } A, B;
      A.h[0] = *(const v8h*)(ap + k0 + 8 * half);
      A.h[1] = *(const v8h*)(ap + k0 + 16 + 8 * half);
      // B 32x16 f16: elems 0..7 -> K = k0+8h..+7 ; elems 8..15 -> K = k0+16+8h..
      B.h[0] = *(const v8h*)(bp + k0 + 8 * half);
      B.h[1] = *(const v8h*)(bp + k0 + 16 + 8 * half);
      acc = __builtin_amdgcn_wmma_f32_16x16x32_f16(
          /*neg_a=*/false, A.v, /*neg_b=*/false, B.v,
          /*c_mod=*/(short)0, acc, /*reuse_a=*/false, /*reuse_b=*/false);
    }
    // D layout: elem r -> (M = r + 8*half, N = l16)
    #pragma unroll
    for (int r = 0; r < 8; ++r) {
      const int mi = i0 + r + 8 * half;
      Cs[mi * C_PITCH + j0 + l16] = 1.0f - acc[r];
    }
  }
  __syncthreads();

  // ---- Phase 2: log-domain Sinkhorn entirely in LDS -----------------------
  const float log_mu = __logf(1.0f / (float)NPG + 1e-8f);  // mu uniform

  for (int it = 0; it < MAX_ITER; ++it) {
    // row pass: u[i] = eps*(log_mu - lse_j S[i,j]) + u[i]
    float errAcc = 0.0f;
    for (int i = wave; i < NPG; i += 8) {
      const float ui = u[i];
      const float* crow = Cs + i * C_PITCH;
      const unsigned* srow = supp + i * SUPP_WPR;
      float s[8];
      float m = -3.0e38f;
      #pragma unroll
      for (int q = 0; q < 8; ++q) {
        const int j = lane + 32 * q;
        const bool on = (srow[q] >> lane) & 1u;
        s[q] = on ? (ui + v[j] - crow[j]) * INV_EPS : NEG_INF;
        m = fmaxf(m, s[q]);
      }
      m = wave_max32(m);
      float sum = 0.0f;
      #pragma unroll
      for (int q = 0; q < 8; ++q) sum += __expf(s[q] - m);
      sum = wave_sum32(sum);
      const float lse = m + __logf(sum);
      const float un = EPS * (log_mu - lse) + ui;
      if (lane == 0) { u[i] = un; errAcc += fabsf(un - ui); }
    }
    if (lane == 0) red[wave] = errAcc;
    __syncthreads();
    float err = 0.0f;
    #pragma unroll
    for (int w = 0; w < 8; ++w) err += red[w];   // same value on all threads

    // col pass: v[j] = eps*(log_mu - lse_i S[i,j]) + v[j]  (uses new u)
    for (int j = wave; j < NPG; j += 8) {
      const float vj = v[j];
      const int wq = j >> 5;
      const unsigned bit = 1u << (j & 31);
      float s[8];
      float m = -3.0e38f;
      #pragma unroll
      for (int q = 0; q < 8; ++q) {
        const int i = lane + 32 * q;
        const bool on = (supp[i * SUPP_WPR + wq] & bit) != 0u;
        s[q] = on ? (u[i] + vj - Cs[i * C_PITCH + j]) * INV_EPS : NEG_INF;
        m = fmaxf(m, s[q]);
      }
      m = wave_max32(m);
      float sum = 0.0f;
      #pragma unroll
      for (int q = 0; q < 8; ++q) sum += __expf(s[q] - m);
      sum = wave_sum32(sum);
      const float lse = m + __logf(sum);
      if (lane == 0) v[j] = EPS * (log_mu - lse) + vj;
    }
    __syncthreads();
    if (err < THRESH) break;   // uniform across the workgroup
  }

  // ---- Phase 3: wd_b = sum_ij pi * C, pi = exp(S) on support --------------
  float acc = 0.0f;
  for (int i = wave; i < NPG; i += 8) {
    const float ui = u[i];
    const float* crow = Cs + i * C_PITCH;
    const unsigned* srow = supp + i * SUPP_WPR;
    #pragma unroll
    for (int q = 0; q < 8; ++q) {
      const int j = lane + 32 * q;
      if ((srow[q] >> lane) & 1u) {
        const float c = crow[j];
        acc += __expf((ui + v[j] - c) * INV_EPS) * c;
      }
    }
  }
  acc = wave_sum32(acc);
  if (lane == 0) red[wave] = acc;
  __syncthreads();
  if (tid == 0) {
    float t = 0.0f;
    #pragma unroll
    for (int w = 0; w < 8; ++w) t += red[w];
    wd_out[b] = t * (0.5f / (float)N_GRAPHS);   // fold 0.5*mean scaling here
  }
}

// ---------------------------------------------------------------------------
// Kernel 5: deterministic fixed-order final reduction (no float atomics)
// ---------------------------------------------------------------------------
__global__ void gtot_final_reduce(const float* __restrict__ wd,
                                  float* __restrict__ out) {
  if (threadIdx.x == 0 && blockIdx.x == 0) {
    float s = 0.0f;
    for (int b = 0; b < N_GRAPHS; ++b) s += wd[b];
    out[0] = s;
  }
}

// ---------------------------------------------------------------------------
// Host launcher
// ---------------------------------------------------------------------------
extern "C" void kernel_launch(void* const* d_in, const int* in_sizes, int n_in,
                              void* d_out, int out_size, void* d_ws, size_t ws_size,
                              hipStream_t stream) {
  (void)in_sizes; (void)n_in; (void)out_size; (void)ws_size;
  const float* fm_src = (const float*)d_in[0];
  const float* fm_tgt = (const float*)d_in[1];
  const int*   eidx   = (const int*)d_in[2];
  // d_in[3] (batch), d_in[4] (n_max=256), d_in[5] (num_graphs=256): structure
  // is fixed (sorted equal-size graphs), sizes baked in as constants.

  char* ws = (char*)d_ws;
  const size_t XS_OFF = 0;
  const size_t XT_OFF = (size_t)N_NODES * D_FEAT * sizeof(_Float16);   // 32 MB
  const size_t SP_OFF = 2 * XT_OFF;                                    // 64 MB
  const size_t WD_OFF = SP_OFF + (size_t)N_GRAPHS * SUPP_WPG * 4;      // +2 MB

  _Float16* xs_h = (_Float16*)(ws + XS_OFF);
  _Float16* xt_h = (_Float16*)(ws + XT_OFF);
  unsigned* supp = (unsigned*)(ws + SP_OFF);
  float*    wd   = (float*)(ws + WD_OFF);
  float*    out  = (float*)d_out;

  // 1) normalize rows -> f16 (one wave per row; 8 rows per 256-thread block)
  gtot_rownorm_f16<<<N_NODES / 8, 256, 0, stream>>>(fm_src, xs_h);
  gtot_rownorm_f16<<<N_NODES / 8, 256, 0, stream>>>(fm_tgt, xt_h);

  // 2) support mask: diagonal init, then edge scatter
  gtot_supp_init<<<(N_GRAPHS * SUPP_WPG) / 256, 256, 0, stream>>>(supp);
  gtot_supp_edges<<<N_EDGES / 256, 256, 0, stream>>>(eidx, supp);

  // 3) per-graph WMMA GEMM + Sinkhorn, all in 273 KB of LDS per workgroup
  gtot_graph<<<N_GRAPHS, 256, LDS_BYTES, stream>>>(xs_h, xt_h, supp, wd);

  // 4) deterministic scalar reduction
  gtot_final_reduce<<<1, 32, 0, stream>>>(wd, out);
}
